// CF_GEN_89670327206619
// MI455X (gfx1250) — matrix operand order repaired and checked by reading, not compiled, not run on hardware
//
#include <hip/hip_runtime.h>
#include <hip/hip_bf16.h>
#include <math.h>

typedef __attribute__((ext_vector_type(16))) _Float16 v16h;
typedef __attribute__((ext_vector_type(8)))  _Float16 v8h;
typedef __attribute__((ext_vector_type(2)))  _Float16 v2h;
typedef __attribute__((ext_vector_type(8)))  float    v8f;
typedef __attribute__((ext_vector_type(4)))  int      v4i;

#define NB   32      // batch
#define NS   512     // sequence length
#define NKW  8       // knows per step
#define ND   256     // hidden dim
#define NH   4       // heads
#define NDK  64      // head dim
#define NEGV (-1.0e32f)

#if defined(__has_builtin)
#if __has_builtin(__builtin_amdgcn_global_load_async_to_lds_b128) && \
    __has_builtin(__builtin_amdgcn_s_wait_asynccnt)
#define USE_ASYNC_LDS 1
#endif
#endif

#ifdef USE_ASYNC_LDS
typedef __attribute__((address_space(1))) void gvoid;
typedef __attribute__((address_space(3))) void lvoid;
typedef __attribute__((address_space(1))) v4i  gv4i;
typedef __attribute__((address_space(3))) v4i  lv4i;
#endif

// ---------------- device helpers (wave32) ----------------
__device__ __forceinline__ float wredmax(float v) {
#pragma unroll
  for (int o = 16; o > 0; o >>= 1) v = fmaxf(v, __shfl_xor(v, o, 32));
  return v;
}
__device__ __forceinline__ float wredsum(float v) {
#pragma unroll
  for (int o = 16; o > 0; o >>= 1) v += __shfl_xor(v, o, 32);
  return v;
}
__device__ __forceinline__ float softplusf(float x) {
  return (x > 20.f) ? x : log1pf(expf(x));
}
__device__ __forceinline__ v16h join16(v8h a, v8h b) {
  v16h r;
#pragma unroll
  for (int j = 0; j < 8; ++j) { r[j] = a[j]; r[j + 8] = b[j]; }
  return r;
}

// ---------------- GEMM: C[M,N] = A[M,K] @ W[K,N] + bias, optional relu ----
// Block = 256 threads (8 waves) computes a 128x64 tile. Per K-step of 32:
// A(128x32 f32) staged via async global->LDS DMA (ASYNCcnt) when available;
// W(32x64) staged transposed as f16. Each wave holds one A fragment
// (f32->f16 cvt at fragment build) and reuses it over 4 WMMAs.
template <int RELU>
__global__ __launch_bounds__(256) void gemm_wmma(
    const float* __restrict__ A, const float* __restrict__ W,
    const float* __restrict__ bias, float* __restrict__ C,
    int M, int N, int Kd)
{
  __shared__ alignas(16) float    sAf[128][32];  // [row][k] f32 (async-DMA'd)
  __shared__ alignas(16) _Float16 sB[64][32];    // [n][k] f16 (transposed)
  const int tid  = threadIdx.x;
  const int lane = tid & 31;
  const int wave = tid >> 5;
  const int lm   = lane & 15;
  const int kha  = (lane < 16) ? 0 : 8;
  const int khb  = (lane < 16) ? 0 : 16;
  const int gn0  = blockIdx.x << 6;   // 64-wide N tile
  const int gm0  = blockIdx.y << 7;   // 128-wide M tile

  v8f acc[4];
#pragma unroll
  for (int nt = 0; nt < 4; ++nt) {
    const float bv = bias[gn0 + (nt << 4) + lm];
#pragma unroll
    for (int r = 0; r < 8; ++r) acc[nt][r] = bv;
  }

  for (int k0 = 0; k0 < Kd; k0 += 32) {
    __syncthreads();   // protect LDS tiles from previous iteration's readers
    // stage A tile: 128x32 floats = 1024 16B segments, 4 per thread
#pragma unroll
    for (int it = 0; it < 4; ++it) {
      const int idx = tid + (it << 8);
      const int r = idx >> 3, seg = (idx & 7) << 2;
      const float* gp = &A[(size_t)(gm0 + r) * Kd + k0 + seg];
#ifdef USE_ASYNC_LDS
      __builtin_amdgcn_global_load_async_to_lds_b128(
          (gv4i*)(gvoid*)(void*)const_cast<float*>(gp),
          (lv4i*)(lvoid*)(void*)&sAf[r][seg], 0, 0);
#else
      *(float4*)&sAf[r][seg] = *(const float4*)gp;
#endif
    }
    // stage W tile transposed+converted: sB[n][k]
    {
      const int bn = tid & 63;
      for (int kk = tid >> 6; kk < 32; kk += 4)
        sB[bn][kk] = (_Float16)W[(size_t)(k0 + kk) * N + gn0 + bn];
    }
#ifdef USE_ASYNC_LDS
    __builtin_amdgcn_s_wait_asynccnt(0);
#endif
    __syncthreads();

    const int arow = (wave << 4) + lm;
    const float* ar = &sAf[arow][kha];
    v16h afr;
#pragma unroll
    for (int j = 0; j < 8; ++j) {
      afr[j]     = (_Float16)ar[j];
      afr[j + 8] = (_Float16)ar[j + 16];
    }
#pragma unroll
    for (int nt = 0; nt < 4; ++nt) {
      const int bn = (nt << 4) + lm;
      const v16h bfr = join16(*(const v8h*)&sB[bn][kha],
                              *(const v8h*)&sB[bn][kha + 16]);
      acc[nt] = __builtin_amdgcn_wmma_f32_16x16x32_f16(
          false, afr, false, bfr, (short)0, acc[nt], false, false);
    }
  }

  const int mb = gm0 + (wave << 4) + ((lane < 16) ? 0 : 8);
#pragma unroll
  for (int nt = 0; nt < 4; ++nt) {
#pragma unroll
    for (int r = 0; r < 8; ++r) {
      float v = acc[nt][r];
      if (RELU) v = fmaxf(v, 0.0f);
      C[(size_t)(mb + r) * N + gn0 + (nt << 4) + lm] = v;
    }
  }
}

// ---------------- attention (one block = one (b,h,16-query tile)) --------
__global__ __launch_bounds__(256) void attn_wmma(
    const float* __restrict__ q, const float* __restrict__ k,
    const float* __restrict__ v, const int* __restrict__ probs,
    const float* __restrict__ gammas, float* __restrict__ o, int fwd)
{
  __shared__ float sp[16][NS + 16];               // scores -> probabilities
  __shared__ alignas(16) _Float16 sq[16][NDK];    // f16 query tile
  __shared__ alignas(16) _Float16 stile[128 * NDK]; // k chunk [key][d] / vT chunk [d][key]
  const int tid  = threadIdx.x;
  const int lane = tid & 31;
  const int wave = tid >> 5;
  const int s0 = blockIdx.x << 4;
  const int h  = blockIdx.y;
  const int bb = blockIdx.z;
  const float* qh = q + (size_t)bb * NS * ND + h * NDK;
  const float* kh = k + (size_t)bb * NS * ND + h * NDK;
  const float* vh = v + (size_t)bb * NS * ND + h * NDK;
  const int lm  = lane & 15;
  const int kha = (lane < 16) ? 0 : 8;
  const int khb = (lane < 16) ? 0 : 16;

  // stage q tile (16x64) as f16
  for (int idx = tid; idx < 16 * 32; idx += 256) {
    const int r = idx >> 5, c2 = idx & 31;
    const float2 qv = *(const float2*)&qh[(size_t)(s0 + r) * ND + (c2 << 1)];
    v2h p; p[0] = (_Float16)qv.x; p[1] = (_Float16)qv.y;
    *(v2h*)&sq[r][c2 << 1] = p;
  }

  // ---- step 1: sc[16][512] = (q_tile @ k^T) / sqrt(DK), 128-key chunks ----
  for (int cc = 0; cc < 4; ++cc) {
    const int c0 = cc << 7;
    __syncthreads();
    // stage k chunk [key][d] as f16 (float2-coalesced)
    {
      const int c2 = tid & 31;
      for (int r = tid >> 5; r < 128; r += 8) {
        const float2 kv = *(const float2*)&kh[(size_t)(c0 + r) * ND + (c2 << 1)];
        v2h p; p[0] = (_Float16)kv.x; p[1] = (_Float16)kv.y;
        *(v2h*)&stile[r * NDK + (c2 << 1)] = p;
      }
    }
    __syncthreads();
    const int n0c = wave << 4;   // this wave's 16-key tile within the chunk
    v8f acc;
#pragma unroll
    for (int r = 0; r < 8; ++r) acc[r] = 0.0f;
#pragma unroll
    for (int k0 = 0; k0 < NDK; k0 += 32) {
      const v16h afr = join16(*(const v8h*)&sq[lm][k0 + kha],
                              *(const v8h*)&sq[lm][k0 + kha + 16]);
      const _Float16* kr = &stile[(n0c + lm) * NDK + k0 + kha];
      const v16h bfr = join16(*(const v8h*)kr, *(const v8h*)(kr + 16));
      acc = __builtin_amdgcn_wmma_f32_16x16x32_f16(
          false, afr, false, bfr, (short)0, acc, false, false);
    }
    const int mb = (lane < 16) ? 0 : 8;
#pragma unroll
    for (int r = 0; r < 8; ++r) sp[mb + r][c0 + n0c + lm] = acc[r] * 0.125f;
  }
  __syncthreads();

  // ---- step 2: distance-decayed double softmax, row-wise (wave per row) --
  const float g = -softplusf(gammas[h]);
  for (int row = wave; row < 16; row += 8) {
    const int i = s0 + row;
    float scl[16], smv[16];
    unsigned mbits = 0;
    float mx = NEGV;
#pragma unroll
    for (int jj = 0; jj < 16; ++jj) {
      const int j = lane * 16 + jj;
      bool m = fwd ? (j <= i) : (j >= i);
      m = m || (probs[(size_t)bb * NS + j] == 0);
      const float s = sp[row][j];
      scl[jj] = s;
      if (m) mbits |= (1u << jj);
      mx = fmaxf(mx, m ? NEGV : s);
    }
    mx = wredmax(mx);
    float lsum = 0.f;
#pragma unroll
    for (int jj = 0; jj < 16; ++jj) {
      const float e = ((mbits >> jj) & 1u) ? 0.f : expf(scl[jj] - mx);
      smv[jj] = e; lsum += e;
    }
    const float tsum = wredsum(lsum);
    const float inv = (tsum > 0.f) ? (1.f / tsum) : 0.f;
    float run = 0.f, pref[16];
#pragma unroll
    for (int jj = 0; jj < 16; ++jj) { smv[jj] *= inv; run += smv[jj]; pref[jj] = run; }
    float scan = run;
#pragma unroll
    for (int o2 = 1; o2 < 32; o2 <<= 1) {
      const float y = __shfl_up(scan, o2, 32);
      if (lane >= o2) scan += y;
    }
    const float excl = scan - run;
    const float dt = __shfl(scan, 31, 32);
    float val[16];
    float mx2 = NEGV;
#pragma unroll
    for (int jj = 0; jj < 16; ++jj) {
      const int j = lane * 16 + jj;
      const float Pinc = excl + pref[jj];
      const float a = fwd ? (dt - Pinc) : (Pinc - smv[jj]);
      const float pos = fabsf((float)(i - j));
      const float dist = sqrtf(fmaxf(a * pos, 0.f));
      const float te = fminf(fmaxf(expf(dist * g), 1e-5f), 1e5f);
      val[jj] = ((mbits >> jj) & 1u) ? NEGV : scl[jj] * te;
      mx2 = fmaxf(mx2, val[jj]);
    }
    mx2 = wredmax(mx2);
    float lsum2 = 0.f;
#pragma unroll
    for (int jj = 0; jj < 16; ++jj) { val[jj] = expf(val[jj] - mx2); lsum2 += val[jj]; }
    const float inv2 = 1.f / wredsum(lsum2);
#pragma unroll
    for (int jj = 0; jj < 16; ++jj) sp[row][lane * 16 + jj] = val[jj] * inv2;
  }

  // ---- step 3: o_tile = p(16x512) @ v(512x64), 128-key chunks ------------
  v8f oacc;
#pragma unroll
  for (int r = 0; r < 8; ++r) oacc[r] = 0.0f;
  for (int cc = 0; cc < 4; ++cc) {
    const int c0 = cc << 7;
    __syncthreads();
    // stage v chunk transposed: stile[d][key] as f16
    for (int idx = tid; idx < 128 * NDK; idx += 256) {
      const int r = idx >> 6;   // key in chunk
      const int c = idx & 63;   // head dim
      stile[c * 128 + r] = (_Float16)vh[(size_t)(c0 + r) * ND + c];
    }
    __syncthreads();
    if (wave < 4) {
      const int n0 = wave << 4;
#pragma unroll
      for (int k0c = 0; k0c < 128; k0c += 32) {
        v16h afr;
#pragma unroll
        for (int j = 0; j < 16; ++j)
          afr[j] = (_Float16)sp[lm][c0 + k0c + kha + ((j < 8) ? j : (j + 8))];
        const _Float16* vr = &stile[(n0 + lm) * 128 + k0c + kha];
        const v16h bfr = join16(*(const v8h*)vr, *(const v8h*)(vr + 16));
        oacc = __builtin_amdgcn_wmma_f32_16x16x32_f16(
            false, afr, false, bfr, (short)0, oacc, false, false);
      }
    }
  }
  if (wave < 4) {
    const int n0 = wave << 4;
    const int mb = (lane < 16) ? 0 : 8;
#pragma unroll
    for (int r = 0; r < 8; ++r)
      o[((size_t)bb * NS + s0 + mb + r) * ND + h * NDK + n0 + lm] = oacc[r];
  }
}

// ---------------- residual add + LayerNorm (wave per row) ----------------
__global__ __launch_bounds__(256) void add_ln(
    const float* __restrict__ X, const float* __restrict__ Y,
    const float* __restrict__ gam, const float* __restrict__ bet,
    float* __restrict__ out)
{
  const int lane = threadIdx.x & 31, wave = threadIdx.x >> 5;
  const size_t row = (size_t)blockIdx.x * 8 + wave;
  const float* xr = X + row * ND;
  const float* yr = Y + row * ND;
  float vals[8];
  float s = 0.f;
#pragma unroll
  for (int e = 0; e < 8; ++e) {
    const int d = lane + e * 32;
    vals[e] = xr[d] + yr[d];
    s += vals[e];
  }
  const float mean = wredsum(s) * (1.0f / ND);
  float vv = 0.f;
#pragma unroll
  for (int e = 0; e < 8; ++e) { const float t = vals[e] - mean; vv += t * t; }
  const float inv = rsqrtf(wredsum(vv) * (1.0f / ND) + 1e-5f);
#pragma unroll
  for (int e = 0; e < 8; ++e) {
    const int d = lane + e * 32;
    out[row * ND + d] = (vals[e] - mean) * inv * gam[d] + bet[d];
  }
}

// ---------------- embeddings ----------------
__global__ void embed_info(const int* __restrict__ probs,
                           const int* __restrict__ knows,
                           const float* __restrict__ knows_emb,
                           const float* __restrict__ probs_emb,
                           float* __restrict__ info)
{
  const int d = threadIdx.x;
  const size_t bs = blockIdx.x;
  const int* kn = knows + bs * NKW;
  float acc = probs_emb[(size_t)probs[bs] * ND + d];
#pragma unroll
  for (int kk = 0; kk < NKW; ++kk)
    acc += knows_emb[(size_t)kn[kk] * ND + d];
  info[bs * ND + d] = acc;
}

__global__ void make_x(const float* __restrict__ info, const int* __restrict__ corrs,
                       const float* __restrict__ te, float* __restrict__ xf,
                       float* __restrict__ xb, int mode)
{
  const int d = threadIdx.x;
  const size_t bs = blockIdx.x;
  const int c = corrs[bs] != 0;
  int t;
  if (mode == 0)      t = c ? 1 : 2;   // right_types
  else if (mode == 1) t = c ? 2 : 0;   // wrong_types
  else                t = c ? 1 : 0;   // types
  const float v = info[bs * ND + d] + te[t * ND + d];
  xf[bs * ND + d] = v;
  xb[bs * ND + d] = v;
}

// ---------------- combine / concat / final predict ----------------
__global__ void combine_k(const float* __restrict__ f, const float* __restrict__ b,
                          float* __restrict__ out)
{
  const int d = threadIdx.x;
  const int s = blockIdx.x, bb = blockIdx.y;
  const size_t base = ((size_t)bb * NS + s) * ND + d;
  float v;
  if (s == 0)            v = b[base];
  else if (s == NS - 1)  v = f[base];
  else                   v = f[base - ND] + b[base + ND];
  out[base] = v;
}

__global__ void cat_k(const float* __restrict__ a, const float* __restrict__ b2,
                      float* __restrict__ out)
{
  const size_t row = blockIdx.x;
  const int c = threadIdx.x;   // 512 threads
  out[row * (2 * ND) + c] = (c < ND) ? a[row * ND + c] : b2[row * ND + (c - ND)];
}

__global__ __launch_bounds__(256) void predict_out(
    const float* __restrict__ h, const float* __restrict__ w3,
    const float* __restrict__ b3, float* __restrict__ out, int t)
{
  const int lane = threadIdx.x & 31, wave = threadIdx.x >> 5;
  const size_t row = (size_t)blockIdx.x * 8 + wave;
  float s = 0.f;
#pragma unroll
  for (int e = 0; e < 4; ++e) {
    const int d = lane + e * 32;
    s += h[row * (ND / 2) + d] * w3[d];
  }
  s = wredsum(s);
  if (lane == 0) {
    const float z = s + b3[0];
    out[row * 3 + t] = 1.f / (1.f + expf(-z));
  }
}

// ---------------- host orchestration ----------------
struct BlkP {
  const float *b1,*b2,*bk,*bo,*bq,*bv,*gam,*ln1b,*ln1g,*ln2b,*ln2g,*w1,*w2,*wk,*wo,*wq,*wv;
};

static BlkP get_blk(void* const* d_in, int base) {
  BlkP p;
  p.b1   = (const float*)d_in[base + 0];
  p.b2   = (const float*)d_in[base + 1];
  p.bk   = (const float*)d_in[base + 2];
  p.bo   = (const float*)d_in[base + 3];
  p.bq   = (const float*)d_in[base + 4];
  p.bv   = (const float*)d_in[base + 5];
  p.gam  = (const float*)d_in[base + 6];
  p.ln1b = (const float*)d_in[base + 7];
  p.ln1g = (const float*)d_in[base + 8];
  p.ln2b = (const float*)d_in[base + 9];
  p.ln2g = (const float*)d_in[base + 10];
  p.w1   = (const float*)d_in[base + 11];
  p.w2   = (const float*)d_in[base + 12];
  p.wk   = (const float*)d_in[base + 13];
  p.wo   = (const float*)d_in[base + 14];
  p.wq   = (const float*)d_in[base + 15];
  p.wv   = (const float*)d_in[base + 16];
  return p;
}

static inline void gemm(const float* A, const float* W, const float* bias, float* C,
                        int M, int N, int K, int relu, hipStream_t s) {
  dim3 grid(N / 64, M / 128);
  if (relu) gemm_wmma<1><<<grid, 256, 0, s>>>(A, W, bias, C, M, N, K);
  else      gemm_wmma<0><<<grid, 256, 0, s>>>(A, W, bias, C, M, N, K);
}

static void run_block(const BlkP& p, float* x, float* qb, float* kb, float* vb,
                      float* ao, float* tmp, float* h1, const int* probs,
                      int fwd, hipStream_t s) {
  const int M = NB * NS;
  gemm(x, p.wq, p.bq, qb, M, ND, ND, 0, s);
  gemm(x, p.wk, p.bk, kb, M, ND, ND, 0, s);
  gemm(x, p.wv, p.bv, vb, M, ND, ND, 0, s);
  attn_wmma<<<dim3(NS / 16, NH, NB), 256, 0, s>>>(qb, kb, vb, probs, p.gam, ao, fwd);
  gemm(ao, p.wo, p.bo, tmp, M, ND, ND, 0, s);
  add_ln<<<M / 8, 256, 0, s>>>(x, tmp, p.ln1g, p.ln1b, x);
  gemm(x, p.w1, p.b1, h1, M, ND, ND, 1, s);
  gemm(h1, p.w2, p.b2, tmp, M, ND, ND, 0, s);
  add_ln<<<M / 8, 256, 0, s>>>(x, tmp, p.ln2g, p.ln2b, x);
}

extern "C" void kernel_launch(void* const* d_in, const int* in_sizes, int n_in,
                              void* d_out, int out_size, void* d_ws, size_t ws_size,
                              hipStream_t stream) {
  const int* probs = (const int*)d_in[0];
  const int* knows = (const int*)d_in[1];
  const int* corrs = (const int*)d_in[2];
  // params pytree-flattened (sorted dict keys):
  // 3..36 b_layers[0..1] (17 leaves each), 37..70 f_layers[0..1],
  // 71 knows_emb, 72 pb1, 73 pb2, 74 pb3, 75 probs_emb, 76 pw1, 77 pw2,
  // 78 pw3, 79 types_emb
  BlkP bl[2] = { get_blk(d_in, 3),  get_blk(d_in, 3 + 17) };
  BlkP fl[2] = { get_blk(d_in, 37), get_blk(d_in, 37 + 17) };
  const float* knows_emb = (const float*)d_in[71];
  const float* pb1 = (const float*)d_in[72];
  const float* pb2 = (const float*)d_in[73];
  const float* pb3 = (const float*)d_in[74];
  const float* probs_emb = (const float*)d_in[75];
  const float* pw1 = (const float*)d_in[76];
  const float* pw2 = (const float*)d_in[77];
  const float* pw3 = (const float*)d_in[78];
  const float* types_emb = (const float*)d_in[79];

  const size_t M = (size_t)NB * NS;
  float* w = (float*)d_ws;
  float* info = w; w += M * ND;
  float* xf   = w; w += M * ND;
  float* xb   = w; w += M * ND;
  float* qb   = w; w += M * ND;
  float* kb   = w; w += M * ND;
  float* vb   = w; w += M * ND;
  float* ao   = w; w += M * ND;
  float* tmp  = w; w += M * ND;
  float* h1   = w; w += M * ND;
  float* comb = w; w += M * ND;
  float* catb = w; w += M * 2 * ND;
  float* ph2  = w; w += M * (ND / 2);
  float* out = (float*)d_out;

  embed_info<<<(unsigned)M, ND, 0, stream>>>(probs, knows, knows_emb, probs_emb, info);

  for (int t = 0; t < 3; ++t) {
    make_x<<<(unsigned)M, ND, 0, stream>>>(info, corrs, types_emb, xf, xb, t);
    for (int l = 0; l < 2; ++l)
      run_block(fl[l], xf, qb, kb, vb, ao, tmp, h1, probs, 1, stream);
    for (int l = 0; l < 2; ++l)
      run_block(bl[l], xb, qb, kb, vb, ao, tmp, h1, probs, 0, stream);
    combine_k<<<dim3(NS, NB), ND, 0, stream>>>(xf, xb, comb);
    cat_k<<<(unsigned)M, 2 * ND, 0, stream>>>(comb, info, catb);
    gemm(catb, pw1, pb1, tmp, (int)M, ND, 2 * ND, 1, stream);
    gemm(tmp, pw2, pb2, ph2, (int)M, ND / 2, ND, 1, stream);
    predict_out<<<(unsigned)(M / 8), 256, 0, stream>>>(ph2, pw3, pb3, out, t);
  }
  (void)in_sizes; (void)n_in; (void)out_size; (void)ws_size;
}
